// HSTU_33775622815820
// MI455X (gfx1250) — matrix operand order repaired and checked by reading, not compiled, not run on hardware
//
#include <hip/hip_runtime.h>
#include <stdint.h>

typedef __attribute__((ext_vector_type(16))) _Float16 v16h;
typedef __attribute__((ext_vector_type(8)))  _Float16 h8;
typedef __attribute__((ext_vector_type(4)))  _Float16 h4;
typedef __attribute__((ext_vector_type(8)))  float    v8f;
typedef __attribute__((ext_vector_type(4)))  float    f4;

#define WMMA_F16(a, b, c) \
  __builtin_amdgcn_wmma_f32_16x16x32_f16(false, (a), false, (b), (short)0, (c), false, false)

// Async global->LDS 128-bit copy (gfx1250, ASYNCcnt).
__device__ __forceinline__ void async_cp16(void* lds, const void* gptr) {
  uint32_t la = (uint32_t)(uintptr_t)lds;  // low 32 bits of flat ptr = LDS address
  asm volatile("global_load_async_to_lds_b128 %0, %1, off" :: "v"(la), "v"(gptr) : "memory");
}
__device__ __forceinline__ void async_wait0() {
  asm volatile("s_wait_asynccnt 0x0" ::: "memory");
}

// ---- WMMA fragment loaders (CDNA5 ISA 7.12.2, wave32) ----
__device__ __forceinline__ v16h frag_a(const _Float16* lds, int ld, int r0, int koff, int lane) {
  int row = r0 + (lane & 15);
  int kb  = koff + ((lane & 16) ? 8 : 0);
  h8 lo = *(const h8*)(lds + row * ld + kb);
  h8 hi = *(const h8*)(lds + row * ld + kb + 16);
  v16h r;
#pragma unroll
  for (int i = 0; i < 8; ++i) { r[i] = lo[i]; r[i + 8] = hi[i]; }
  return r;
}
__device__ __forceinline__ v16h frag_b(const _Float16* lds, int ld, int n0, int koff, int lane) {
  int col = n0 + (lane & 15);
  int kb  = koff + ((lane & 16) ? 8 : 0);
  h8 lo = *(const h8*)(lds + col * ld + kb);
  h8 hi = *(const h8*)(lds + col * ld + kb + 16);
  v16h r;
#pragma unroll
  for (int i = 0; i < 8; ++i) { r[i] = lo[i]; r[i + 8] = hi[i]; }
  return r;
}

// 16 WMMAs on a 128x128 tile (per 64-K step); wave = 32x64 sub-tile.
__device__ __forceinline__ void compute_tile(const _Float16* As, const _Float16* Bs,
                                             v8f* acc, int wm, int wn, int lane) {
  v16h a[4];
#pragma unroll
  for (int ms = 0; ms < 2; ++ms)
#pragma unroll
    for (int kh = 0; kh < 2; ++kh)
      a[ms * 2 + kh] = frag_a(As, 64, wm * 32 + ms * 16, kh * 32, lane);
#pragma unroll
  for (int ns = 0; ns < 4; ++ns)
#pragma unroll
    for (int kh = 0; kh < 2; ++kh) {
      v16h bf = frag_b(Bs, 64, wn * 64 + ns * 16, kh * 32, lane);
      acc[ns]     = WMMA_F16(a[kh],     bf, acc[ns]);
      acc[4 + ns] = WMMA_F16(a[2 + kh], bf, acc[4 + ns]);
    }
}

// Store staged registers: A f32x4 -> f16 LDS [m][k]; B f32x4 -> f16 LDS [n][k] (transposed).
__device__ __forceinline__ void stage_store(_Float16* As, _Float16* Bs, const f4* ra, const f4* rb,
                                            int ar, int ac, int bk, int bn) {
#pragma unroll
  for (int j = 0; j < 8; ++j) {
    h4 h; h[0] = (_Float16)ra[j][0]; h[1] = (_Float16)ra[j][1];
    h[2] = (_Float16)ra[j][2]; h[3] = (_Float16)ra[j][3];
    *(h4*)&As[(ar + 16 * j) * 64 + ac] = h;
  }
#pragma unroll
  for (int j = 0; j < 8; ++j)
#pragma unroll
    for (int t = 0; t < 4; ++t) Bs[(bn + t) * 64 + bk + 8 * j] = (_Float16)rb[j][t];
}

// ---- Kernel 1: proj = x @ Wp + bp ; scatter U (f32) and Q/K/V (f16 head-major) ----
__global__ __launch_bounds__(256) void hstu_gemm_proj(
    const float* __restrict__ x, const float* __restrict__ Wp, const float* __restrict__ bp,
    float* __restrict__ U, _Float16* __restrict__ Qh, _Float16* __restrict__ Kh,
    _Float16* __restrict__ Vh) {
  __shared__ __align__(16) _Float16 As[2][128 * 64];
  __shared__ __align__(16) _Float16 Bs[2][128 * 64];
  const int tid = threadIdx.x, lane = tid & 31, wid = tid >> 5;
  const int wm = wid >> 1, wn = wid & 1;
  const int m0 = blockIdx.y * 128, n0 = blockIdx.x * 128;
  const int ar = tid >> 4, ac = (tid & 15) << 2;      // A stage coords
  const int bk = tid >> 5, bn = (tid & 31) << 2;      // B stage coords
  const float* xp = x  + (size_t)(m0 + ar) * 1024 + ac;
  const float* wp = Wp + (size_t)bk * 4096 + n0 + bn;
  v8f acc[8] = {};
  f4 ra[8], rb[8];
  // Prologue: gather + stage tile 0.
#pragma unroll
  for (int j = 0; j < 8; ++j) ra[j] = *(const f4*)(xp + (size_t)(16 * j) * 1024);
#pragma unroll
  for (int j = 0; j < 8; ++j) rb[j] = *(const f4*)(wp + (size_t)(8 * j) * 4096);
  stage_store(As[0], Bs[0], ra, rb, ar, ac, bk, bn);
  __syncthreads();
  for (int k0 = 0; k0 < 1024; k0 += 64) {
    const int pb = (k0 >> 6) & 1;
    const bool more = (k0 + 64) < 1024;
    if (more) {  // gather next tile while this tile computes
#pragma unroll
      for (int j = 0; j < 8; ++j) ra[j] = *(const f4*)(xp + (size_t)(16 * j) * 1024 + k0 + 64);
#pragma unroll
      for (int j = 0; j < 8; ++j) rb[j] = *(const f4*)(wp + (size_t)(k0 + 64 + 8 * j) * 4096);
      __builtin_prefetch(xp + (size_t)(16 * (tid & 7)) * 1024 + k0 + 128, 0, 3);
      __builtin_prefetch(wp + (size_t)(k0 + 128 + 8 * (tid & 7)) * 4096, 0, 3);
    }
    compute_tile(As[pb], Bs[pb], acc, wm, wn, lane);
    if (more) stage_store(As[pb ^ 1], Bs[pb ^ 1], ra, rb, ar, ac, bk, bn);
    __syncthreads();
  }
  const int hi8 = (lane & 16) ? 8 : 0, nc = lane & 15;
#pragma unroll
  for (int ms = 0; ms < 2; ++ms)
#pragma unroll
    for (int ns = 0; ns < 4; ++ns) {
      v8f c = acc[ms * 4 + ns];
#pragma unroll
      for (int r = 0; r < 8; ++r) {
        int grow = m0 + wm * 32 + ms * 16 + r + hi8;
        int gcol = n0 + wn * 64 + ns * 16 + nc;
        float val = c[r] + bp[gcol];
        int seg = gcol >> 10, cc = gcol & 1023;
        int b = grow >> 11, s = grow & 2047;
        int head = cc >> 6, d = cc & 63;
        size_t hidx = (((size_t)(b * 16 + head)) * 2048 + s) * 64 + d;
        if (seg == 0)      U[(size_t)grow * 1024 + cc] = val;
        else if (seg == 1) Vh[hidx] = (_Float16)val;
        else if (seg == 2) Qh[hidx] = (_Float16)val;
        else               Kh[hidx] = (_Float16)val;
      }
    }
}

// ---- Kernel 2: in-place RoPE on Qh/Kh ----
__global__ __launch_bounds__(128) void hstu_rope(_Float16* __restrict__ Qh,
                                                 _Float16* __restrict__ Kh) {
  const int t = threadIdx.x & 31;
  const int row = blockIdx.x * 4 + (threadIdx.x >> 5);
  _Float16* buf = (blockIdx.y == 0 ? Qh : Kh) + (size_t)row * 64;
  const int s = row & 2047;
  float inv = __powf(10000.0f, -(float)(2 * t) * (1.0f / 64.0f));
  float ang = (float)s * inv;
  float cv = __cosf(ang), sv = __sinf(ang);
  float x1 = (float)buf[t], x2 = (float)buf[t + 32];
  buf[t]      = (_Float16)(x1 * cv - x2 * sv);
  buf[t + 32] = (_Float16)(x2 * cv + x1 * sv);
}

// ---- Kernel 3: causal sigmoid attention, double-buffered k-tiles ----
__global__ __launch_bounds__(128) void hstu_attn(
    const _Float16* __restrict__ Qh, const _Float16* __restrict__ Kh,
    const _Float16* __restrict__ Vh, float* __restrict__ AO) {
  __shared__ __align__(16) _Float16 Qs[64 * 64];        // [q][d]
  __shared__ __align__(16) _Float16 Ks[2][32 * 64];     // [k][d]
  __shared__ __align__(16) _Float16 Vt[2][64 * 32];     // [d][k]
  __shared__ __align__(16) _Float16 St[4 * 16 * 32];    // per-wave attn staging
  const int tid = threadIdx.x, lane = tid & 31, wid = tid >> 5;
  const int bh = blockIdx.y;
  const int q0 = blockIdx.x * 64;
  const size_t base = (size_t)bh * 2048 * 64;
  const int vk = tid >> 3, vd = (tid & 7) << 3;         // V stage coords
  // Q tile: async copies (4 x b128 per thread).
#pragma unroll
  for (int i = tid; i < 512; i += 128)
    async_cp16(Qs + i * 8, Qh + base + (size_t)q0 * 64 + i * 8);
  // Prologue k-tile 0: K async, V gather+transpose.
  h8 rv[2];
#pragma unroll
  for (int j = 0; j < 2; ++j) {
    async_cp16(Ks[0] + (tid + 128 * j) * 8, Kh + base + (size_t)(tid + 128 * j) * 8);
    rv[j] = *(const h8*)&Vh[base + (size_t)(vk + 16 * j) * 64 + vd];
  }
#pragma unroll
  for (int j = 0; j < 2; ++j)
#pragma unroll
    for (int t = 0; t < 8; ++t) Vt[0][(vd + t) * 32 + vk + 16 * j] = rv[j][t];
  async_wait0();
  __syncthreads();
  v16h aq0 = frag_a(Qs, 64, wid * 16, 0, lane);
  v16h aq1 = frag_a(Qs, 64, wid * 16, 32, lane);
  v8f acc[4] = {};
  const int hi8 = (lane & 16) ? 8 : 0, nc = lane & 15;
  const int qrow = q0 + wid * 16 + hi8;
  const int nkt = q0 / 32 + 2;  // causal
  _Float16* st = St + wid * (16 * 32);
  for (int kt = 0; kt < nkt; ++kt) {
    const int k0 = kt * 32;
    const int pb = kt & 1;
    const bool more = (kt + 1) < nkt;
    if (more) {  // fetch next k-tile during compute
      const size_t nb = base + (size_t)(k0 + 32) * 64;
#pragma unroll
      for (int j = 0; j < 2; ++j) {
        async_cp16(Ks[pb ^ 1] + (tid + 128 * j) * 8, Kh + nb + (size_t)(tid + 128 * j) * 8);
        rv[j] = *(const h8*)&Vh[nb + (size_t)(vk + 16 * j) * 64 + vd];
      }
    }
    // scores: Q(16x64) x K^T
    v8f s0 = {}, s1 = {};
    s0 = WMMA_F16(aq0, frag_b(Ks[pb], 64, 0, 0, lane), s0);
    s0 = WMMA_F16(aq1, frag_b(Ks[pb], 64, 0, 32, lane), s0);
    s1 = WMMA_F16(aq0, frag_b(Ks[pb], 64, 16, 0, lane), s1);
    s1 = WMMA_F16(aq1, frag_b(Ks[pb], 64, 16, 32, lane), s1);
#pragma unroll
    for (int r = 0; r < 8; ++r) {
      int q = qrow + r;
      int k = k0 + nc;
      float p0 = (k      <= q) ? 1.0f / (1.0f + __expf(-s0[r] * 0.125f)) : 0.0f;
      float p1 = (k + 16 <= q) ? 1.0f / (1.0f + __expf(-s1[r] * 0.125f)) : 0.0f;
      st[(r + hi8) * 32 + nc]      = (_Float16)p0;
      st[(r + hi8) * 32 + nc + 16] = (_Float16)p1;
    }
    __syncthreads();
    v16h ap = frag_a(st, 32, 0, 0, lane);
#pragma unroll
    for (int ns = 0; ns < 4; ++ns)
      acc[ns] = WMMA_F16(ap, frag_b(Vt[pb], 32, ns * 16, 0, lane), acc[ns]);
    if (more) {
#pragma unroll
      for (int j = 0; j < 2; ++j)
#pragma unroll
        for (int t = 0; t < 8; ++t) Vt[pb ^ 1][(vd + t) * 32 + vk + 16 * j] = rv[j][t];
    }
    async_wait0();
    __syncthreads();
  }
  const int b = bh >> 4, h = bh & 15;
#pragma unroll
  for (int ns = 0; ns < 4; ++ns)
#pragma unroll
    for (int r = 0; r < 8; ++r) {
      int s = qrow + r;
      AO[((size_t)b * 2048 + s) * 1024 + h * 64 + ns * 16 + nc] = acc[ns][r];
    }
}

// ---- Kernel 4: layernorm(attn_out) * silu(U) -> f16 gated ----
__global__ __launch_bounds__(256) void hstu_ln_gate(
    const float* __restrict__ AO, const float* __restrict__ U,
    const float* __restrict__ g, const float* __restrict__ bb, _Float16* __restrict__ G) {
  __shared__ float red[256];
  __shared__ float red2[256];
  const int row = blockIdx.x, tid = threadIdx.x;
  const float* a = AO + (size_t)row * 1024;
  f4 v = *(const f4*)&a[tid * 4];
  float s  = v[0] + v[1] + v[2] + v[3];
  float s2 = v[0]*v[0] + v[1]*v[1] + v[2]*v[2] + v[3]*v[3];
  red[tid] = s; red2[tid] = s2;
  __syncthreads();
  for (int off = 128; off > 0; off >>= 1) {
    if (tid < off) { red[tid] += red[tid + off]; red2[tid] += red2[tid + off]; }
    __syncthreads();
  }
  float mean = red[0] * (1.0f / 1024.0f);
  float var  = red2[0] * (1.0f / 1024.0f) - mean * mean;
  float rstd = rsqrtf(var + 1e-8f);
  f4 uv = *(const f4*)&U[(size_t)row * 1024 + tid * 4];
  f4 gv = *(const f4*)&g[tid * 4];
  f4 bv = *(const f4*)&bb[tid * 4];
  h4 o;
#pragma unroll
  for (int j = 0; j < 4; ++j) {
    float nrm  = (v[j] - mean) * rstd * gv[j] + bv[j];
    float silu = uv[j] / (1.0f + __expf(-uv[j]));
    o[j] = (_Float16)(silu * nrm);
  }
  *(h4*)&G[(size_t)row * 1024 + tid * 4] = o;
}

// ---- Kernel 5: out = x + gated @ Wt + bt (A tile async f16, B staged f32->f16) ----
__global__ __launch_bounds__(256) void hstu_gemm_out(
    const _Float16* __restrict__ G, const float* __restrict__ Wt,
    const float* __restrict__ bt, const float* __restrict__ x, float* __restrict__ out) {
  __shared__ __align__(16) _Float16 As[2][128 * 64];
  __shared__ __align__(16) _Float16 Bs[2][128 * 64];
  const int tid = threadIdx.x, lane = tid & 31, wid = tid >> 5;
  const int wm = wid >> 1, wn = wid & 1;
  const int m0 = blockIdx.y * 128, n0 = blockIdx.x * 128;
  const int ar = tid >> 3, ac = (tid & 7) << 3;       // A async coords (h8 units)
  const int bk = tid >> 5, bn = (tid & 31) << 2;
  const _Float16* gp = G + (size_t)(m0 + ar) * 1024 + ac;
  const float* wp = Wt + (size_t)bk * 1024 + n0 + bn;
  v8f acc[8] = {};
  f4 rb[8];
  // Prologue tile 0: async A, gather B.
#pragma unroll
  for (int j = 0; j < 4; ++j)
    async_cp16(&As[0][(ar + 32 * j) * 64 + ac], gp + (size_t)(32 * j) * 1024);
#pragma unroll
  for (int j = 0; j < 8; ++j) rb[j] = *(const f4*)(wp + (size_t)(8 * j) * 1024);
#pragma unroll
  for (int j = 0; j < 8; ++j)
#pragma unroll
    for (int t = 0; t < 4; ++t) Bs[0][(bn + t) * 64 + bk + 8 * j] = (_Float16)rb[j][t];
  async_wait0();
  __syncthreads();
  for (int k0 = 0; k0 < 1024; k0 += 64) {
    const int pb = (k0 >> 6) & 1;
    const bool more = (k0 + 64) < 1024;
    if (more) {
#pragma unroll
      for (int j = 0; j < 4; ++j)
        async_cp16(&As[pb ^ 1][(ar + 32 * j) * 64 + ac], gp + (size_t)(32 * j) * 1024 + k0 + 64);
#pragma unroll
      for (int j = 0; j < 8; ++j) rb[j] = *(const f4*)(wp + (size_t)(k0 + 64 + 8 * j) * 1024);
      __builtin_prefetch(wp + (size_t)(k0 + 128 + 8 * (tid & 7)) * 1024, 0, 3);
    }
    compute_tile(As[pb], Bs[pb], acc, wm, wn, lane);
    if (more) {
#pragma unroll
      for (int j = 0; j < 8; ++j)
#pragma unroll
        for (int t = 0; t < 4; ++t) Bs[pb ^ 1][(bn + t) * 64 + bk + 8 * j] = (_Float16)rb[j][t];
    }
    async_wait0();
    __syncthreads();
  }
  const int hi8 = (lane & 16) ? 8 : 0, nc = lane & 15;
#pragma unroll
  for (int ms = 0; ms < 2; ++ms)
#pragma unroll
    for (int ns = 0; ns < 4; ++ns) {
      v8f c = acc[ms * 4 + ns];
#pragma unroll
      for (int r = 0; r < 8; ++r) {
        int grow = m0 + wm * 32 + ms * 16 + r + hi8;
        int gcol = n0 + wn * 64 + ns * 16 + nc;
        out[(size_t)grow * 1024 + gcol] =
            x[(size_t)grow * 1024 + gcol] + c[r] + bt[gcol];
      }
    }
}

extern "C" void kernel_launch(void* const* d_in, const int* in_sizes, int n_in,
                              void* d_out, int out_size, void* d_ws, size_t ws_size,
                              hipStream_t stream) {
  (void)in_sizes; (void)n_in; (void)out_size; (void)ws_size;
  const float* x    = (const float*)d_in[0];
  const float* Wp   = (const float*)d_in[2];
  const float* bp   = (const float*)d_in[3];
  const float* ln_g = (const float*)d_in[4];
  const float* ln_b = (const float*)d_in[5];
  const float* Wt   = (const float*)d_in[6];
  const float* bt   = (const float*)d_in[7];

  char* ws = (char*)d_ws;
  float*    U   = (float*)   (ws);                                  // 32 MB
  _Float16* Qh  = (_Float16*)(ws + 33554432);                       // 16 MB
  _Float16* Kh  = (_Float16*)(ws + 33554432 + 16777216);            // 16 MB
  _Float16* Vh  = (_Float16*)(ws + 33554432 + 2 * 16777216);        // 16 MB
  float*    AO  = (float*)   (ws + 33554432 + 3 * 16777216);        // 32 MB
  _Float16* G   = (_Float16*)(ws + 2 * 33554432 + 3 * 16777216);    // 16 MB
  float*    out = (float*)d_out;

  hstu_gemm_proj<<<dim3(32, 64), 256, 0, stream>>>(x, Wp, bp, U, Qh, Kh, Vh);
  hstu_rope<<<dim3(32768, 2), 128, 0, stream>>>(Qh, Kh);
  hstu_attn<<<dim3(32, 64), 128, 0, stream>>>(Qh, Kh, Vh, AO);
  hstu_ln_gate<<<8192, 256, 0, stream>>>(AO, U, ln_g, ln_b, G);
  hstu_gemm_out<<<dim3(8, 64), 256, 0, stream>>>(G, Wt, bt, x, out);
}
